// MultiHeadSelfAttention_79413945303077
// MI455X (gfx1250) — matrix operand order repaired and checked by reading, not compiled, not run on hardware
//
#include <hip/hip_runtime.h>
#include <hip/hip_bf16.h>
#include <stdint.h>
#include <stddef.h>

// ---------------------------------------------------------------------------
// Types matching gfx1250 WMMA operand shapes (wave32).
// ---------------------------------------------------------------------------
typedef __bf16 bf16;
typedef __attribute__((ext_vector_type(16))) __bf16 bf16x16;
typedef __attribute__((ext_vector_type(8)))  __bf16 bf16x8;
typedef __attribute__((ext_vector_type(8)))  float  f32x8;

#define WMMA_BF16(a, b, c) \
  __builtin_amdgcn_wmma_f32_16x16x32_bf16(false, (a), false, (b), (short)0, (c), false, false)

// A-matrix 16x32 bf16 fragment: lane holds row M=lane%16, 16 consecutive K
// values starting at k0 (+16 for lanes 16-31), matching load_b_frag.
// Source must be row-major [row][ldk] bf16.
__device__ __forceinline__ bf16x16 load_a_frag(const bf16* base, int ldk, int row,
                                               int k0, int lane) {
  return *reinterpret_cast<const bf16x16*>(base + (size_t)row * ldk + k0 + ((lane >= 16) ? 16 : 0));
}

// B-matrix 32x16 bf16 fragment: lane holds column N=lane%16, 16 consecutive K
// values starting at k0 (+16 for lanes 16-31). Source stored K-contiguous per
// column (i.e. weight rows (N,K) row-major).
__device__ __forceinline__ bf16x16 load_b_frag(const bf16* base, int ldk, int col,
                                               int k0, int lane) {
  return *reinterpret_cast<const bf16x16*>(base + (size_t)col * ldk + k0 + ((lane >= 16) ? 16 : 0));
}

// ---------------------------------------------------------------------------
// Kernel 0: fp32 -> bf16 conversion of x and weights; zero LN stats.
// ---------------------------------------------------------------------------
__global__ __launch_bounds__(256) void convert_kernel(
    const float* __restrict__ x, const float* __restrict__ wqk,
    const float* __restrict__ wv, const float* __restrict__ wo,
    bf16* __restrict__ xb, bf16* __restrict__ wqkvb, bf16* __restrict__ wob,
    float* __restrict__ stats) {
  const size_t NX = 8388608ull, NQK = 2097152ull, NV = 1048576ull, NO = 1048576ull;
  size_t idx = (size_t)blockIdx.x * blockDim.x + threadIdx.x;
  if (idx < NX) {
    xb[idx] = (bf16)x[idx];
  } else if (idx < NX + NQK) {
    size_t i = idx - NX;
    wqkvb[i] = (bf16)wqk[i];
  } else if (idx < NX + NQK + NV) {
    size_t i = idx - NX;                // rows 2048..3071 of combined weight
    wqkvb[i] = (bf16)wv[i - NQK];
  } else if (idx < NX + NQK + NV + NO) {
    size_t i = idx - NX - NQK - NV;
    wob[i] = (bf16)wo[i];
  }
  if (idx < 16) stats[idx] = 0.0f;      // per-batch sum/sumsq accumulators
}

// ---------------------------------------------------------------------------
// Kernel 1: fused QKV projection. M=8192 (B*S), K=1024, N=3072.
// One 64x64 output tile per wave (4x4 WMMA tiles, K-loop step 32):
// 16 b128 loads per 16 WMMAs -> 32 FLOP per L2 byte of operand traffic.
// Epilogue adds bias and scatters into attention-friendly layouts:
//   Q[b][h][s][kd], K[b][h][s][kd]  (kd contiguous)
//   Vt[b][h][vd][s]                 (s contiguous, B-frag ready for PV)
// ---------------------------------------------------------------------------
__global__ __launch_bounds__(256) void qkv_gemm_kernel(
    const bf16* __restrict__ xb, const bf16* __restrict__ wqkv,
    const float* __restrict__ b_qk, const float* __restrict__ b_v,
    bf16* __restrict__ Q, bf16* __restrict__ K, bf16* __restrict__ Vt) {
  const int lane = threadIdx.x & 31;
  const int wid  = (int)((blockIdx.x * blockDim.x + threadIdx.x) >> 5);
  const int TN = 3072 / 64;  // 48 tile-columns
  const int tm = (wid / TN) * 64;
  const int tn = (wid % TN) * 64;
  const int rbase = lane & 15;
  f32x8 acc[4][4] = {};
  for (int k0 = 0; k0 < 1024; k0 += 32) {
    bf16x16 A[4], Bf[4];
#pragma unroll
    for (int i = 0; i < 4; ++i)
      A[i] = load_a_frag(xb, 1024, tm + i * 16 + rbase, k0, lane);
#pragma unroll
    for (int j = 0; j < 4; ++j)
      Bf[j] = load_b_frag(wqkv, 1024, tn + j * 16 + rbase, k0, lane);
#pragma unroll
    for (int i = 0; i < 4; ++i)
#pragma unroll
      for (int j = 0; j < 4; ++j)
        acc[i][j] = WMMA_BF16(A[i], Bf[j], acc[i][j]);
  }
#pragma unroll
  for (int mi = 0; mi < 4; ++mi)
#pragma unroll
    for (int ni = 0; ni < 4; ++ni)
#pragma unroll
      for (int r = 0; r < 8; ++r) {
        int row = tm + mi * 16 + ((lane >> 4) << 3) + r;   // C/D layout
        int col = tn + ni * 16 + (lane & 15);
        int b = row >> 10, s = row & 1023;
        float v = acc[mi][ni][r];
        if (col < 2048) {
          v += b_qk[col];
          int kd = col >> 5, rem = col & 31, h = rem & 15;  // c = kd*32+sel*16+h
          bf16* dst = (rem < 16) ? Q : K;
          dst[(((size_t)(b * 16 + h)) * 1024 + s) * 64 + kd] = (bf16)v;
        } else {
          int cv = col - 2048;
          v += b_v[cv];
          int vd = cv >> 4, h = cv & 15;                    // c = vd*16+h
          Vt[(((size_t)(b * 16 + h)) * 64 + vd) * 1024 + s] = (bf16)v;
        }
      }
}

// ---------------------------------------------------------------------------
// Kernel 2: flash attention per (b,h). 1024 blocks = 128 bh * 8 row-blocks.
// Each of 8 waves owns 16 query rows; online softmax over j in chunks of 32.
// P is staged through per-wave LDS to convert C/D layout -> A-frag layout.
// ---------------------------------------------------------------------------
__global__ __launch_bounds__(256) void attn_kernel(
    const bf16* __restrict__ Q, const bf16* __restrict__ K,
    const bf16* __restrict__ Vt, bf16* __restrict__ AO) {
  __shared__ bf16 pshare[8][16 * 32];
  const int lane = threadIdx.x & 31;
  const int wave = threadIdx.x >> 5;
  const int bh = blockIdx.x >> 3;                   // b*16 + h
  const int i0 = (blockIdx.x & 7) * 128 + wave * 16;
  const int b = bh >> 4, h = bh & 15;
  const bf16* Qb = Q  + (size_t)bh * 1024 * 64;
  const bf16* Kb = K  + (size_t)bh * 1024 * 64;
  const bf16* Vb = Vt + (size_t)bh * 64 * 1024;

  bf16x16 qa0 = load_a_frag(Qb, 64, i0 + (lane & 15), 0,  lane);
  bf16x16 qa1 = load_a_frag(Qb, 64, i0 + (lane & 15), 32, lane);

  f32x8 o[4] = {};
  float mrow[8], lrow[8];
#pragma unroll
  for (int r = 0; r < 8; ++r) { mrow[r] = -1e30f; lrow[r] = 0.0f; }

  for (int j0 = 0; j0 < 1024; j0 += 32) {
    // scores S = Q K^T for 16 rows x 32 keys (KD=64 -> two K-steps each)
    f32x8 s0 = {}, s1 = {};
    bf16x16 k00 = load_b_frag(Kb, 64, j0 + (lane & 15),      0,  lane);
    bf16x16 k01 = load_b_frag(Kb, 64, j0 + (lane & 15),      32, lane);
    bf16x16 k10 = load_b_frag(Kb, 64, j0 + 16 + (lane & 15), 0,  lane);
    bf16x16 k11 = load_b_frag(Kb, 64, j0 + 16 + (lane & 15), 32, lane);
    s0 = WMMA_BF16(qa0, k00, s0);  s0 = WMMA_BF16(qa1, k01, s0);
    s1 = WMMA_BF16(qa0, k10, s1);  s1 = WMMA_BF16(qa1, k11, s1);

    // online softmax (row reductions within 16-lane halves via shfl_xor)
#pragma unroll
    for (int r = 0; r < 8; ++r) {
      float v0 = s0[r] * 0.125f, v1 = s1[r] * 0.125f;   // 1/sqrt(64)
      float mx = fmaxf(v0, v1);
#pragma unroll
      for (int off = 1; off < 16; off <<= 1) mx = fmaxf(mx, __shfl_xor(mx, off, 32));
      float mnew = fmaxf(mrow[r], mx);
      float fac = __expf(mrow[r] - mnew);
      float p0 = __expf(v0 - mnew), p1 = __expf(v1 - mnew);
      float sum = p0 + p1;
#pragma unroll
      for (int off = 1; off < 16; off <<= 1) sum += __shfl_xor(sum, off, 32);
      lrow[r] = lrow[r] * fac + sum;
      mrow[r] = mnew;
#pragma unroll
      for (int f4 = 0; f4 < 4; ++f4) o[f4][r] *= fac;
      int rl = ((lane >> 4) << 3) + r;                 // C/D row this elem holds
      pshare[wave][rl * 32 + (lane & 15)]      = (bf16)p0;
      pshare[wave][rl * 32 + 16 + (lane & 15)] = (bf16)p1;
    }
    // reload P in A-frag layout (in-order DS within the wave; no barrier)
    bf16x16 pa = load_a_frag(&pshare[wave][0], 32, lane & 15, 0, lane);
#pragma unroll
    for (int ni = 0; ni < 4; ++ni) {
      bf16x16 vb = load_b_frag(Vb, 1024, ni * 16 + (lane & 15), j0, lane);
      o[ni] = WMMA_BF16(pa, vb, o[ni]);
    }
  }
  // normalize and store as bf16 in [b*S + i][h*64 + vd] (A-ready for O-proj)
#pragma unroll
  for (int ni = 0; ni < 4; ++ni)
#pragma unroll
    for (int r = 0; r < 8; ++r) {
      int i = i0 + ((lane >> 4) << 3) + r;
      int vd = ni * 16 + (lane & 15);
      float val = o[ni][r] / lrow[r];
      AO[((size_t)(b * 1024 + i)) * 1024 + h * 64 + vd] = (bf16)val;
    }
}

// ---------------------------------------------------------------------------
// Kernel 3: output projection + bias + residual; accumulate per-batch LN stats.
// M=8192, N=1024, K=1024; 64x64 tile per wave.
// ---------------------------------------------------------------------------
__global__ __launch_bounds__(256) void out_proj_kernel(
    const bf16* __restrict__ AO, const bf16* __restrict__ Wo,
    const float* __restrict__ b_o, const float* __restrict__ x,
    float* __restrict__ ybuf, float* __restrict__ stats) {
  const int lane = threadIdx.x & 31;
  const int wid  = (int)((blockIdx.x * blockDim.x + threadIdx.x) >> 5);
  const int TN = 1024 / 64;  // 16
  const int tm = (wid / TN) * 64;
  const int tn = (wid % TN) * 64;
  const int rbase = lane & 15;
  f32x8 acc[4][4] = {};
  for (int k0 = 0; k0 < 1024; k0 += 32) {
    bf16x16 A[4], Bf[4];
#pragma unroll
    for (int i = 0; i < 4; ++i)
      A[i] = load_a_frag(AO, 1024, tm + i * 16 + rbase, k0, lane);
#pragma unroll
    for (int j = 0; j < 4; ++j)
      Bf[j] = load_b_frag(Wo, 1024, tn + j * 16 + rbase, k0, lane);
#pragma unroll
    for (int i = 0; i < 4; ++i)
#pragma unroll
      for (int j = 0; j < 4; ++j)
        acc[i][j] = WMMA_BF16(A[i], Bf[j], acc[i][j]);
  }
  float lsum = 0.0f, lsq = 0.0f;
#pragma unroll
  for (int mi = 0; mi < 4; ++mi)
#pragma unroll
    for (int ni = 0; ni < 4; ++ni)
#pragma unroll
      for (int r = 0; r < 8; ++r) {
        int row = tm + mi * 16 + ((lane >> 4) << 3) + r;
        int col = tn + ni * 16 + (lane & 15);
        float v = acc[mi][ni][r] + b_o[col] + x[(size_t)row * 1024 + col];
        ybuf[(size_t)row * 1024 + col] = v;
        lsum += v;
        lsq += v * v;
      }
#pragma unroll
  for (int off = 1; off < 32; off <<= 1) {
    lsum += __shfl_xor(lsum, off, 32);
    lsq  += __shfl_xor(lsq,  off, 32);
  }
  if (lane == 0) {
    int bidx = tm >> 10;  // whole 64-row tile lies in one batch (1024 % 64 == 0)
    atomicAdd(&stats[bidx * 2 + 0], lsum);
    atomicAdd(&stats[bidx * 2 + 1], lsq);
  }
}

// ---------------------------------------------------------------------------
// Kernel 4: per-batch LayerNorm apply (mean/var over S*D = 2^20 elements).
// ---------------------------------------------------------------------------
__global__ __launch_bounds__(256) void ln_kernel(
    const float* __restrict__ ybuf, const float* __restrict__ stats,
    const float* __restrict__ ln_w, const float* __restrict__ ln_b,
    float* __restrict__ out) {
  size_t idx = (size_t)blockIdx.x * blockDim.x + threadIdx.x;  // < 8*2^20
  int b = (int)(idx >> 20);
  size_t sd = idx & 1048575ull;
  const float inv_n = 1.0f / 1048576.0f;
  float mean = stats[b * 2 + 0] * inv_n;
  float var  = stats[b * 2 + 1] * inv_n - mean * mean;
  float rs = rsqrtf(var + 1e-5f);
  float v = (ybuf[idx] - mean) * rs;
  out[idx] = v * ln_w[sd] + ln_b[sd];
}

// ---------------------------------------------------------------------------
// Launch
// ---------------------------------------------------------------------------
extern "C" void kernel_launch(void* const* d_in, const int* in_sizes, int n_in,
                              void* d_out, int out_size, void* d_ws, size_t ws_size,
                              hipStream_t stream) {
  const float* x    = (const float*)d_in[0];
  const float* W_qk = (const float*)d_in[1];
  const float* b_qk = (const float*)d_in[2];
  const float* W_v  = (const float*)d_in[3];
  const float* b_v  = (const float*)d_in[4];
  const float* W_o  = (const float*)d_in[5];
  const float* b_o  = (const float*)d_in[6];
  const float* ln_w = (const float*)d_in[7];
  const float* ln_b = (const float*)d_in[8];
  float* out = (float*)d_out;

  char* ws = (char*)d_ws;
  size_t off = 0;
  auto alloc = [&](size_t bytes) -> void* {
    void* p = ws + off;
    off += (bytes + 255) & ~(size_t)255;
    return p;
  };
  bf16*  xb    = (bf16*)alloc(8388608ull * 2);              // x bf16        16 MB
  bf16*  wqkvb = (bf16*)alloc(3145728ull * 2);              // [Wqk;Wv] bf16  6 MB
  bf16*  wob   = (bf16*)alloc(1048576ull * 2);              // Wo bf16        2 MB
  bf16*  Qb    = (bf16*)alloc(8ull * 16 * 1024 * 64 * 2);   // Q             16 MB
  bf16*  Kb    = (bf16*)alloc(8ull * 16 * 1024 * 64 * 2);   // K             16 MB
  bf16*  Vtb   = (bf16*)alloc(8ull * 16 * 64 * 1024 * 2);   // V^T           16 MB
  bf16*  AO    = (bf16*)alloc(8388608ull * 2);              // attn out      16 MB
  float* ybuf  = (float*)alloc(8388608ull * 4);             // pre-LN y      32 MB
  float* stats = (float*)alloc(64);                         // per-batch sums

  convert_kernel<<<49152, 256, 0, stream>>>(x, W_qk, W_v, W_o, xb, wqkvb, wob, stats);
  qkv_gemm_kernel<<<768, 256, 0, stream>>>(xb, wqkvb, b_qk, b_v, Qb, Kb, Vtb);
  attn_kernel<<<1024, 256, 0, stream>>>(Qb, Kb, Vtb, AO);
  out_proj_kernel<<<256, 256, 0, stream>>>(AO, wob, b_o, x, ybuf, stats);
  ln_kernel<<<32768, 256, 0, stream>>>(ybuf, stats, ln_w, ln_b, out);
}
